// GNNEdgePredictor_59923383714097
// MI455X (gfx1250) — matrix operand order repaired and checked by reading, not compiled, not run on hardware
//
#include <hip/hip_runtime.h>

// ---------------------------------------------------------------------------
// GNN edge predictor for MI455X (gfx1250, wave32, WMMA).
// Dominant cost: edge MLP ~426 GFLOP on v_wmma_f32_16x16x32_f16.
// Key optimization this round: weights staged cooperatively into LDS
// (one copy per block instead of per wave) => L2 weight traffic cut 4x,
// B-fragments served from LDS; A fragments pinned in VGPRs (128 VGPRs/wave).
// ---------------------------------------------------------------------------

#define HH   256     // hidden channels
#define HH2  512     // 2*H
#define LL   2       // GNN layers
#define BN_EPS 1e-5f

typedef _Float16 f16;
typedef _Float16 v16h __attribute__((ext_vector_type(16)));
typedef float    v8f  __attribute__((ext_vector_type(8)));

union Frag { v16h v; float4 f4[2]; };

// ------------------------------- utility kernels ---------------------------

__global__ void k_f32_to_f16(const float* __restrict__ s, f16* __restrict__ d, long n) {
  long i = (long)blockIdx.x * blockDim.x + threadIdx.x;
  if (i < n) d[i] = (f16)s[i];
}

__global__ void k_fill_zero(float* __restrict__ p, long n) {
  long i = (long)blockIdx.x * blockDim.x + threadIdx.x;
  if (i < n) p[i] = 0.0f;
}

// Bt[n*K + k] = W[k*Nout + n]  (f32 row-major [K][N] -> f16 [N][K])
__global__ void k_pack_bt(const float* __restrict__ W, f16* __restrict__ Bt,
                          int K, int Nout) {
  int i = blockIdx.x * blockDim.x + threadIdx.x;
  if (i < K * Nout) {
    int n = i / K, k = i - n * K;
    Bt[i] = (f16)W[(long)k * Nout + n];
  }
}

__global__ void k_deg(const int* __restrict__ dst, float* __restrict__ deg, int E) {
  int i = blockIdx.x * blockDim.x + threadIdx.x;
  if (i < E) atomicAdd(&deg[dst[i]], 1.0f);
}

__global__ void k_dis(float* __restrict__ degdis, int N) {
  int i = blockIdx.x * blockDim.x + threadIdx.x;
  if (i < N) degdis[i] = rsqrtf(degdis[i] + 1.0f);  // +1 = self loop
}

// ---------------------------- conv GEMM: xw = xh @ W -----------------------
// 8 waves/block, each wave one 16-row tile, 16 n-tiles, K=256 (8 wmma steps).
// Weights are tiny (128KB) and L2-resident; A gets register-cached by LICM.
__global__ __launch_bounds__(256)
void k_conv_gemm(const f16* __restrict__ xh, const f16* __restrict__ Bt,
                 float* __restrict__ xw, int Nnodes) {
  int wid = threadIdx.x >> 5, lane = threadIdx.x & 31;
  int row0 = (blockIdx.x * 8 + wid) * 16;
  if (row0 >= Nnodes) return;
  int r = lane & 15, kh = lane >> 4;
  const f16* ap0 = xh + (long)(row0 + r) * HH + kh * 8;
  for (int nt = 0; nt < 16; ++nt) {
    int n = nt * 16 + r;
    const f16* bp0 = Bt + (long)n * HH + kh * 8;
    v8f acc;
#pragma unroll
    for (int j = 0; j < 8; ++j) acc[j] = 0.0f;
#pragma unroll
    for (int k0 = 0; k0 < HH; k0 += 32) {
      Frag a, b;
      a.f4[0] = *(const float4*)(ap0 + k0);
      a.f4[1] = *(const float4*)(ap0 + k0 + 16);
      b.f4[0] = *(const float4*)(bp0 + k0);
      b.f4[1] = *(const float4*)(bp0 + k0 + 16);
      acc = __builtin_amdgcn_wmma_f32_16x16x32_f16(false, a.v, false, b.v,
                                                   (short)0, acc, false, false);
    }
#pragma unroll
    for (int j = 0; j < 8; ++j)
      xw[(long)(row0 + kh * 8 + j) * HH + n] = acc[j];
  }
}

// ------------------------- message passing (scatter) -----------------------
__global__ __launch_bounds__(256)
void k_scatter(const int* __restrict__ src, const int* __restrict__ dst,
               const float* __restrict__ dis, const float* __restrict__ xw,
               float* __restrict__ y, int E) {
  int wid = threadIdx.x >> 5, lane = threadIdx.x & 31;
  long e = (long)blockIdx.x * 8 + wid;
  if (e >= E) return;
  int s = src[e], d = dst[e];
  float c = dis[s] * dis[d];
  const float* xs = xw + (long)s * HH + lane * 8;
  float* yd = y + (long)d * HH + lane * 8;
#pragma unroll
  for (int j = 0; j < 8; ++j) atomicAdd(yd + j, c * xs[j]);
}

// y += dis[n]^2 * xw + convB[h]
__global__ void k_selfloop_bias(const float* __restrict__ dis,
                                const float* __restrict__ xw,
                                const float* __restrict__ convB,
                                float* __restrict__ y, long NH) {
  long i = (long)blockIdx.x * blockDim.x + threadIdx.x;
  if (i < NH) {
    int n = (int)(i >> 8), h = (int)(i & 255);
    float c = dis[n];
    y[i] += c * c * xw[i] + convB[h];
  }
}

// ------------------------------- batch norm --------------------------------
__global__ __launch_bounds__(256)
void k_bn_stats(const float* __restrict__ y, float* __restrict__ stats, int N) {
  int h = threadIdx.x;
  float s = 0.0f, s2 = 0.0f;
  for (int n = blockIdx.x; n < N; n += gridDim.x) {
    float v = y[(long)n * HH + h];
    s += v; s2 += v * v;
  }
  atomicAdd(&stats[h], s);
  atomicAdd(&stats[HH + h], s2);
}

__global__ void k_bn_apply(const float* __restrict__ y, const float* __restrict__ stats,
                           const float* __restrict__ gamma, const float* __restrict__ beta,
                           f16* __restrict__ xh, int N) {
  long i = (long)blockIdx.x * blockDim.x + threadIdx.x;
  long NH = (long)N * HH;
  if (i >= NH) return;
  int h = (int)(i & 255);
  float inv = 1.0f / (float)N;
  float mu = stats[h] * inv;
  float var = stats[HH + h] * inv - mu * mu;
  float v = (y[i] - mu) * rsqrtf(var + BN_EPS) * gamma[h] + beta[h];
  xh[i] = (f16)(v > 0.0f ? v : 0.0f);
}

// ----------------------- fused edge MLP (WMMA pipeline) --------------------
// Block = 128 threads = 4 waves = 64 edges. Per-block LDS (128KB):
//   h1: 64x512 f16 (64KB)  [offset 0]       (later overlaid by h3: 64x128)
//   Bs: 16384 f16  (32KB)  [offset 32768]   weight staging slab
//   h2: 64x256 f16 (32KB)  [offset 49152]
// => 2 blocks/WGP so staging barriers of one block overlap the other's WMMAs.

__device__ __forceinline__
void stage_weights(const f16* __restrict__ gsrc, f16* __restrict__ Bs,
                   int n_f16, int tid, int nthr) {
  const uint4* s4 = (const uint4*)gsrc;
  uint4* d4 = (uint4*)Bs;
  for (int i = tid; i < (n_f16 >> 3); i += nthr) d4[i] = s4[i];
}

// A fragments pinned in registers; B staged in LDS group-by-group.
template<int K, int NOUT, int LDO>
__device__ __forceinline__
void gemm_regA_stagedB(const Frag* a, const f16* __restrict__ Wt,
                       const float* __restrict__ bias,
                       f16* __restrict__ Bs, f16* __restrict__ outLds,
                       int wid, int lane, int tid) {
  constexpr int ROWS_PER_GROUP = 16384 / K;          // Bs capacity in n-rows
  constexpr int NT_PER_GROUP   = ROWS_PER_GROUP / 16;
  constexpr int NGROUPS        = NOUT / ROWS_PER_GROUP;
  const int r = lane & 15, kh = lane >> 4;
  for (int g = 0; g < NGROUPS; ++g) {
    const int n0 = g * ROWS_PER_GROUP;
    __syncthreads();                                  // Bs free of prior readers
    stage_weights(Wt + (long)n0 * K, Bs, ROWS_PER_GROUP * K, tid, 128);
    __syncthreads();                                  // Bs ready
#pragma unroll
    for (int nt = 0; nt < NT_PER_GROUP; ++nt) {
      const int nl = nt * 16 + r;                     // row within staged slab
      const int n  = n0 + nl;
      float bv = bias[n];
      v8f acc;
#pragma unroll
      for (int j = 0; j < 8; ++j) acc[j] = bv;
      const f16* bp0 = Bs + (long)nl * K + kh * 8;
#pragma unroll
      for (int ks = 0; ks < K / 32; ++ks) {
        Frag b;
        b.f4[0] = *(const float4*)(bp0 + ks * 32);
        b.f4[1] = *(const float4*)(bp0 + ks * 32 + 16);
        acc = __builtin_amdgcn_wmma_f32_16x16x32_f16(false, a[ks].v, false, b.v,
                                                     (short)0, acc, false, false);
      }
#pragma unroll
      for (int j = 0; j < 8; ++j) {
        float v = acc[j];
        v = v > 0.0f ? v : 0.0f;
        outLds[(wid * 16 + kh * 8 + j) * LDO + n] = (f16)v;
      }
    }
  }
}

__device__ __forceinline__
void load_a_global(Frag* a, const f16* __restrict__ srow,
                   const f16* __restrict__ drow, int kh) {
#pragma unroll
  for (int ks = 0; ks < 16; ++ks) {
    const int k0 = ks * 32;
    const f16* xrow = (k0 < HH) ? srow : drow;        // concat [x[src], x[dst]]
    const int col = (k0 & (HH - 1)) + kh * 8;
    a[ks].f4[0] = *(const float4*)(xrow + col);
    a[ks].f4[1] = *(const float4*)(xrow + col + 16);
  }
}

template<int NKS>
__device__ __forceinline__
void load_a_lds(Frag* a, const f16* __restrict__ base) {
#pragma unroll
  for (int ks = 0; ks < NKS; ++ks) {
    a[ks].f4[0] = *(const float4*)(base + ks * 32);
    a[ks].f4[1] = *(const float4*)(base + ks * 32 + 16);
  }
}

__global__ __launch_bounds__(128)
void k_mlp(const int* __restrict__ srcIdx, const int* __restrict__ dstIdx,
           const f16* __restrict__ xh,
           const f16* __restrict__ W1t, const float* __restrict__ b1,
           const f16* __restrict__ W2t, const float* __restrict__ b2,
           const f16* __restrict__ W3t, const float* __restrict__ b3,
           const f16* __restrict__ W4h, const float* __restrict__ b4,
           float* __restrict__ out, int E) {
  extern __shared__ __align__(16) f16 smem[];
  f16* h1 = smem;               // 64 x 512
  f16* Bs = smem + 32768;       // 16384 f16 staging
  f16* h2 = smem + 49152;       // 64 x 256
  f16* h3 = smem;               // overlay: 64 x 128 (h1 dead by GEMM3)

  const int tid = threadIdx.x;
  const int wid = tid >> 5, lane = tid & 31;
  const int r = lane & 15, kh = lane >> 4;

  const long e0 = (long)blockIdx.x * 64;
  long ce = e0 + wid * 16 + r; if (ce >= E) ce = E - 1;
  const int ns = srcIdx[ce], nd = dstIdx[ce];

  Frag a[16];                   // 128 VGPRs of A, reused across all 3 GEMMs

  // GEMM1: edge_emb[64x512] @ W1t -> h1 (ReLU f16)
  load_a_global(a, xh + (long)ns * HH, xh + (long)nd * HH, kh);
  gemm_regA_stagedB<512, 512, 512>(a, W1t, b1, Bs, h1, wid, lane, tid);

  // GEMM2: h1 @ W2t -> h2 (A reloaded from own LDS rows; wave-private)
  load_a_lds<16>(a, h1 + (wid * 16 + r) * 512 + kh * 8);
  gemm_regA_stagedB<512, 256, 256>(a, W2t, b2, Bs, h2, wid, lane, tid);

  // GEMM3: h2 @ W3t -> h3 (overlays h1; safe after gemm3's staging barrier)
  load_a_lds<8>(a, h2 + (wid * 16 + r) * 256 + kh * 8);
  gemm_regA_stagedB<256, 128, 128>(a, W3t, b3, Bs, h3, wid, lane, tid);

  // Final: h3[16x128] . W4 + b4 (own wave's rows; lanes 0..15)
  if (lane < 16) {
    long e = e0 + wid * 16 + lane;
    if (e < E) {
      const f16* h = h3 + (wid * 16 + lane) * 128;
      float s = 0.0f;
#pragma unroll 8
      for (int k = 0; k < 128; ++k) s += (float)h[k] * (float)W4h[k];
      out[e] = s + b4[0];
    }
  }
}

// ------------------------------- host launcher -----------------------------

extern "C" void kernel_launch(void* const* d_in, const int* in_sizes, int n_in,
                              void* d_out, int out_size, void* d_ws, size_t ws_size,
                              hipStream_t stream) {
  (void)n_in; (void)out_size; (void)ws_size;

  const int E = in_sizes[0] / 2;
  const int N = in_sizes[1] / HH;
  const long NH = (long)N * HH;

  const int*   edge_index = (const int*)d_in[0];
  const int*   src = edge_index;
  const int*   dst = edge_index + E;
  const float* node_emb = (const float*)d_in[1];
  const float* convW    = (const float*)d_in[2];
  const float* convB    = (const float*)d_in[3];
  const float* bn_gamma = (const float*)d_in[4];
  const float* bn_beta  = (const float*)d_in[5];
  const float* W1 = (const float*)d_in[6];  const float* b1 = (const float*)d_in[7];
  const float* W2 = (const float*)d_in[8];  const float* b2 = (const float*)d_in[9];
  const float* W3 = (const float*)d_in[10]; const float* b3 = (const float*)d_in[11];
  const float* W4 = (const float*)d_in[12]; const float* b4 = (const float*)d_in[13];
  float* out = (float*)d_out;

  // workspace carving (256B aligned)
  uintptr_t p = (uintptr_t)d_ws;
  auto alloc = [&](size_t bytes) -> void* {
    void* q = (void*)p; p += (bytes + 255) & ~(size_t)255; return q;
  };
  f16*   xh     = (f16*)  alloc((size_t)NH * 2);
  float* xw     = (float*)alloc((size_t)NH * 4);
  float* y      = (float*)alloc((size_t)NH * 4);
  float* dis    = (float*)alloc((size_t)N * 4);
  float* stats  = (float*)alloc(2 * HH * 4);
  f16*   convWt = (f16*)  alloc((size_t)LL * HH * HH * 2);
  f16*   W1t    = (f16*)  alloc((size_t)HH2 * HH2 * 2);
  f16*   W2t    = (f16*)  alloc((size_t)HH * HH2 * 2);
  f16*   W3t    = (f16*)  alloc((size_t)128 * HH * 2);
  f16*   W4h    = (f16*)  alloc(128 * 2);

  // allow 128KB dynamic LDS for the MLP kernel (host-side, capture-safe)
  (void)hipFuncSetAttribute((const void*)k_mlp,
                            hipFuncAttributeMaxDynamicSharedMemorySize, 131072);

  // ---- pack weights to f16 WMMA-B layout (~1.2MB, L2 resident) ----
  k_pack_bt<<<(HH * HH + 255) / 256, 256, 0, stream>>>(convW,            convWt,           HH,  HH);
  k_pack_bt<<<(HH * HH + 255) / 256, 256, 0, stream>>>(convW + HH * HH,  convWt + HH * HH, HH,  HH);
  k_pack_bt<<<(HH2 * HH2 + 255) / 256, 256, 0, stream>>>(W1, W1t, HH2, HH2);
  k_pack_bt<<<(HH2 * HH  + 255) / 256, 256, 0, stream>>>(W2, W2t, HH2, HH);
  k_pack_bt<<<(HH  * 128 + 255) / 256, 256, 0, stream>>>(W3, W3t, HH,  128);
  k_f32_to_f16<<<1, 128, 0, stream>>>(W4, W4h, 128);

  // ---- node features -> f16 ; GCN normalization coefficients ----
  k_f32_to_f16<<<(int)((NH + 255) / 256), 256, 0, stream>>>(node_emb, xh, NH);
  k_fill_zero<<<(N + 255) / 256, 256, 0, stream>>>(dis, N);
  k_deg<<<(E + 255) / 256, 256, 0, stream>>>(dst, dis, E);
  k_dis<<<(N + 255) / 256, 256, 0, stream>>>(dis, N);

  // ---- 2 GCN layers ----
  const int mtiles = (N + 15) / 16;
  for (int l = 0; l < LL; ++l) {
    k_conv_gemm<<<(mtiles + 7) / 8, 256, 0, stream>>>(xh, convWt + (size_t)l * HH * HH, xw, N);
    k_fill_zero<<<(int)((NH + 255) / 256), 256, 0, stream>>>(y, NH);
    k_scatter<<<(E + 7) / 8, 256, 0, stream>>>(src, dst, dis, xw, y, E);
    k_selfloop_bias<<<(int)((NH + 255) / 256), 256, 0, stream>>>(dis, xw, convB + l * HH, y, NH);
    k_fill_zero<<<2, 256, 0, stream>>>(stats, 2 * HH);
    k_bn_stats<<<512, 256, 0, stream>>>(y, stats, N);
    k_bn_apply<<<(int)((NH + 255) / 256), 256, 0, stream>>>(y, stats, bn_gamma + l * HH,
                                                            bn_beta + l * HH, xh, N);
  }

  // ---- fused edge MLP (dominant ~426 GFLOP on WMMA, LDS-staged weights) ----
  k_mlp<<<(E + 63) / 64, 128, 131072, stream>>>(
      src, dst, xh, W1t, b1, W2t, b2, W3t, b3, W4h, b4, out, E);
}
